// ResBlock_14937896256226
// MI455X (gfx1250) — matrix operand order repaired and checked by reading, not compiled, not run on hardware
//
#include <hip/hip_runtime.h>
#include <stdint.h>
#include <stddef.h>

// ---------------- problem constants ----------------
#define NN  200000          // nodes
#define KT  27              // taps
#define CH  64              // channels
#define FRAG_DWORDS (KT*2*4*32*8)   // 55296 dwords = 216 KiB of packed bf16 B-fragments
#define STAT_BLOCKS 512

typedef __attribute__((ext_vector_type(16))) __bf16 v16bf;
typedef __attribute__((ext_vector_type(8)))  float  v8f;

union FragAB { v16bf v; uint4 q[2]; };

__device__ __forceinline__ uint32_t f2bf(float f) {
  union { float f; uint32_t u; } v; v.f = f;
  return (v.u + 0x7FFFu + ((v.u >> 16) & 1u)) >> 16;   // RNE truncate to bf16
}

// ---------------- W packing: [27,64,64] f32 -> WMMA bf16 B-fragment layout ----------------
// fragment f = ((k*2+h)*4 + j); within fragment: lane l owns 8 dwords (32B) contiguous.
// dword r of lane l packs (K = (l<16?0:8)+2(r&3)+16(r>>2) (+c0), col = n0 + (l&15)) pair (lo,hi).
__global__ void __launch_bounds__(256) pack_w_kernel(const float* __restrict__ W,
                                                     uint32_t* __restrict__ frag) {
  int i = blockIdx.x * 256 + threadIdx.x;
  if (i >= FRAG_DWORDS) return;
  int r  = i & 7;
  int l  = (i >> 3) & 31;
  int f  = i >> 8;
  int j  = f & 3;
  int kk = f >> 2;
  int h  = kk & 1;
  int k  = kk >> 1;
  int c0 = 32 * h, n0 = 16 * j;
  int kk0 = ((l < 16) ? 0 : 8) + 2 * (r & 3) + 16 * (r >> 2);
  int col = l & 15;
  float lo = W[(k * CH + c0 + kk0)     * CH + n0 + col];
  float hi = W[(k * CH + c0 + kk0 + 1) * CH + n0 + col];
  frag[i] = f2bf(lo) | (f2bf(hi) << 16);
}

// ---------------- f32 -> bf16 row conversion (2 elems / dword) ----------------
__global__ void __launch_bounds__(256) cvt_f32_bf16(const float* __restrict__ src,
                                                    uint32_t* __restrict__ dst, int n2) {
  for (int i = blockIdx.x * blockDim.x + threadIdx.x; i < n2; i += gridDim.x * blockDim.x) {
    dst[i] = f2bf(src[2 * i]) | (f2bf(src[2 * i + 1]) << 16);
  }
}

// A-fragment gather: both 32-wide K chunks (h=0,1) for two 16-row m-tiles.
// Lane layout matches ISA 16-bit A 16x32: lanes<16 take K 0-7 & 16-23, lanes>=16 take 8-15 & 24-31.
__device__ __forceinline__ void load_a(FragAB a[2][2], const uint32_t* __restrict__ xb,
                                       int i0, int i1, int sel) {
  const uint4* p0 = (const uint4*)(xb + (size_t)i0 * 32) + sel;
  const uint4* p1 = (const uint4*)(xb + (size_t)i1 * 32) + sel;
#pragma unroll
  for (int h = 0; h < 2; ++h) {
    a[h][0].q[0] = p0[h * 4];  a[h][0].q[1] = p0[h * 4 + 2];
    a[h][1].q[0] = p1[h * 4];  a[h][1].q[1] = p1[h * 4 + 2];
  }
}

// ---------------- octree conv: y[n,d] = sum_{k,c} xb[neigh[n,k],c] * Wfrag[k,c,d] ----------------
// 8 waves/WG; wave owns 32 nodes x 64 out-channels. All packed W lives in LDS (216 KiB).
// Two-stage software pipeline: indices fetched one tap ahead, A gathers double-buffered
// so L2 gather latency overlaps the 16 WMMAs of the previous tap.
__global__ void __launch_bounds__(256) octconv_wmma(const uint32_t* __restrict__ xb,   // [NN,32] dwords (bf16 pairs)
                                                    const int* __restrict__ neigh,    // [NN,27]
                                                    const uint32_t* __restrict__ wfrag,
                                                    float* __restrict__ y) {          // [NN,64] f32
  extern __shared__ uint4 ldsW4[];            // FRAG_DWORDS/4 = 13824 uint4
  const int tid  = threadIdx.x;
  const int lane = tid & 31;
  const int wave = tid >> 5;

  const uint4* g4 = (const uint4*)wfrag;
  for (int u = tid; u < FRAG_DWORDS / 4; u += 256) ldsW4[u] = g4[u];
  __syncthreads();

  const int n0    = blockIdx.x * 256 + wave * 32;
  const int sel   = lane >> 4;      // A-layout half: K 0-7/16-23 vs 8-15/24-31
  const int mlane = lane & 15;      // A/B row-col lane id

  int r0 = n0 + mlane;        if (r0 >= NN) r0 = NN - 1;   // clamp (no EXEC divergence)
  int r1 = n0 + 16 + mlane;   if (r1 >= NN) r1 = NN - 1;
  const int base0 = r0 * KT;
  const int base1 = r1 * KT;

  v8f acc[2][4];
#pragma unroll
  for (int t = 0; t < 2; ++t)
#pragma unroll
    for (int j = 0; j < 4; ++j) acc[t][j] = (v8f){0.f,0.f,0.f,0.f,0.f,0.f,0.f,0.f};

  // pipeline prologue: indices + A fragments for tap 0, indices for tap 1
  FragAB abuf[2][2][2];                     // [stage][h][m-tile]
  int i0 = neigh[base0 + 0];
  int i1 = neigh[base1 + 0];
  load_a(abuf[0], xb, i0, i1, sel);
  int ni0 = neigh[base0 + 1];
  int ni1 = neigh[base1 + 1];

#pragma unroll 2
  for (int k = 0; k < KT; ++k) {
    const int cur = k & 1, nxt = cur ^ 1;
    // issue next tap's A gathers (indices landed during previous tap's WMMAs)
    load_a(abuf[nxt], xb, ni0, ni1, sel);
    // fetch indices two taps ahead (clamped; redundant loads at the tail are harmless)
    const int kn2 = (k + 2 < KT) ? (k + 2) : (KT - 1);
    ni0 = neigh[base0 + kn2];
    ni1 = neigh[base1 + kn2];

#pragma unroll
    for (int h = 0; h < 2; ++h) {                 // two 32-wide K chunks per tap
      const int kk = k * 2 + h;
#pragma unroll
      for (int j = 0; j < 4; ++j) {               // 4 output n-tiles
        FragAB b;
        const uint4* lp = ldsW4 + ((size_t)(kk * 4 + j) * 32 + lane) * 2;
        b.q[0] = lp[0];  b.q[1] = lp[1];
        acc[0][j] = __builtin_amdgcn_wmma_f32_16x16x32_bf16(false, abuf[cur][h][0].v, false, b.v,
                                                            (short)0, acc[0][j], false, false);
        acc[1][j] = __builtin_amdgcn_wmma_f32_16x16x32_bf16(false, abuf[cur][h][1].v, false, b.v,
                                                            (short)0, acc[1][j], false, false);
      }
    }
  }

  // C layout: lane 0-15 vgpr v -> row v; lane 16-31 -> row v+8; col = j*16 + (lane&15)
  const int rowsel = (lane >= 16) ? 8 : 0;
#pragma unroll
  for (int t = 0; t < 2; ++t)
#pragma unroll
    for (int v = 0; v < 8; ++v) {
      const int row = n0 + t * 16 + v + rowsel;
      if (row < NN) {
#pragma unroll
        for (int j = 0; j < 4; ++j)
          y[(size_t)row * CH + j * 16 + mlane] = acc[t][j][v];
      }
    }
}

// ---------------- BN stats: deterministic block partials (64 sums + 64 sumsqs per block) ----------------
__global__ void __launch_bounds__(256) bn_stats(const float* __restrict__ y,
                                                float* __restrict__ part) {  // [STAT_BLOCKS,128]
  __shared__ float ls[256], ls2[256];
  const int tid = threadIdx.x;
  const int c   = tid & 63;
  const int rg  = tid >> 6;                    // 4 row groups / block
  float s = 0.f, s2 = 0.f;
  for (int row = blockIdx.x * 4 + rg; row < NN; row += gridDim.x * 4) {
    float v = y[(size_t)row * CH + c];
    s += v; s2 += v * v;
  }
  ls[tid] = s; ls2[tid] = s2;
  __syncthreads();
  if (tid < 128) { ls[tid] += ls[tid + 128]; ls2[tid] += ls2[tid + 128]; }
  __syncthreads();
  if (tid < 64) {
    part[(size_t)blockIdx.x * 128 + tid]      = ls[tid]  + ls[tid + 64];
    part[(size_t)blockIdx.x * 128 + 64 + tid] = ls2[tid] + ls2[tid + 64];
  }
}

// reduce partials -> scale/shift: ss[c]=gamma*rsqrt(var+eps), ss[64+c]=beta-mean*scale
__global__ void __launch_bounds__(128) bn_finalize(const float* __restrict__ part,
                                                   const float* __restrict__ gamma,
                                                   const float* __restrict__ beta,
                                                   float* __restrict__ ss) {
  __shared__ float tot[128];
  const int t = threadIdx.x;
  float s = 0.f;
  for (int b = 0; b < STAT_BLOCKS; ++b) s += part[(size_t)b * 128 + t];
  tot[t] = s;
  __syncthreads();
  if (t < 64) {
    const float invN = 1.0f / (float)NN;
    float mean = tot[t] * invN;
    float var  = tot[64 + t] * invN - mean * mean;
    float sc   = gamma[t] * rsqrtf(var + 1e-5f);
    ss[t]      = sc;
    ss[64 + t] = beta[t] - mean * sc;
  }
}

// BN + ReLU + convert to bf16 (intermediate activation)
__global__ void __launch_bounds__(256) bn_relu_cvt(const float* __restrict__ y,
                                                   const float* __restrict__ ss,
                                                   uint32_t* __restrict__ zb) {
  const int total = NN * (CH / 2);
  for (int i = blockIdx.x * blockDim.x + threadIdx.x; i < total; i += gridDim.x * blockDim.x) {
    const int c = (i & 31) * 2;
    float a = fmaxf(y[2 * i]     * ss[c]     + ss[64 + c],     0.f);
    float b = fmaxf(y[2 * i + 1] * ss[c + 1] + ss[64 + c + 1], 0.f);
    zb[i] = f2bf(a) | (f2bf(b) << 16);
  }
}

// BN + residual + ReLU -> d_out, plus depth passthrough
__global__ void __launch_bounds__(256) final_out(const float* __restrict__ y,
                                                 const float* __restrict__ ss,
                                                 const float* __restrict__ x,
                                                 float* __restrict__ out,
                                                 const int* __restrict__ depth,
                                                 int out_size) {
  const int total = NN * CH;
  const int gid = blockIdx.x * blockDim.x + threadIdx.x;
  for (int i = gid; i < total; i += gridDim.x * blockDim.x) {
    const int c = i & 63;
    out[i] = fmaxf(y[i] * ss[c] + ss[64 + c] + x[i], 0.f);
  }
  if (gid == 0 && out_size > total) ((int*)out)[total] = depth[0];
}

// ---------------- host launch ----------------
extern "C" void kernel_launch(void* const* d_in, const int* in_sizes, int n_in,
                              void* d_out, int out_size, void* d_ws, size_t ws_size,
                              hipStream_t stream) {
  const float* x      = (const float*)d_in[0];
  const int*   neigh  = (const int*)  d_in[1];
  const int*   depth  = (const int*)  d_in[2];
  const float* W1     = (const float*)d_in[3];
  const float* gamma1 = (const float*)d_in[4];
  const float* beta1  = (const float*)d_in[5];
  const float* W2     = (const float*)d_in[6];
  const float* gamma2 = (const float*)d_in[7];
  const float* beta2  = (const float*)d_in[8];
  float* out = (float*)d_out;

  // workspace carve-up (byte offsets, 16B-aligned)
  char* ws = (char*)d_ws;
  uint32_t* xb    = (uint32_t*)(ws);                               // 25,600,000 B (reused as z1b)
  float*    y     = (float*)   (ws + 25600000);                    // 51,200,000 B
  uint32_t* frag1 = (uint32_t*)(ws + 76800000);                    //    221,184 B
  uint32_t* frag2 = (uint32_t*)(ws + 77021184);                    //    221,184 B
  float*    part  = (float*)   (ws + 77242368);                    //    262,144 B
  float*    ss1   = (float*)   (ws + 77504512);                    //        512 B
  float*    ss2   = (float*)   (ws + 77505024);                    //        512 B

  const int convGrid = (NN + 255) / 256;                 // 782 WGs, 8 waves each
  const size_t ldsBytes = (size_t)FRAG_DWORDS * 4;       // 216 KiB dynamic LDS

  // 0) pack weights into WMMA bf16 B-fragment layout
  pack_w_kernel<<<(FRAG_DWORDS + 255) / 256, 256, 0, stream>>>(W1, frag1);
  pack_w_kernel<<<(FRAG_DWORDS + 255) / 256, 256, 0, stream>>>(W2, frag2);

  // 1) x -> bf16
  cvt_f32_bf16<<<4096, 256, 0, stream>>>(x, xb, NN * (CH / 2));

  // 2) conv1 (WMMA), BN1 stats, BN1+ReLU -> bf16 activation (reuse xb buffer)
  octconv_wmma<<<convGrid, 256, ldsBytes, stream>>>(xb, neigh, frag1, y);
  bn_stats<<<STAT_BLOCKS, 256, 0, stream>>>(y, part);
  bn_finalize<<<1, 128, 0, stream>>>(part, gamma1, beta1, ss1);
  bn_relu_cvt<<<4096, 256, 0, stream>>>(y, ss1, xb);

  // 3) conv2 (WMMA), BN2 stats, BN2 + residual + ReLU -> out
  octconv_wmma<<<convGrid, 256, ldsBytes, stream>>>(xb, neigh, frag2, y);
  bn_stats<<<STAT_BLOCKS, 256, 0, stream>>>(y, part);
  bn_finalize<<<1, 128, 0, stream>>>(part, gamma2, beta2, ss2);
  final_out<<<4096, 256, 0, stream>>>(y, ss2, x, out, depth, out_size);
}
